// FeatureFusionBlock_v3_75479755260685
// MI455X (gfx1250) — compile-verified, matrix operand order checked by reading, not run on hardware
//
#include <hip/hip_runtime.h>
#include <hip/hip_bf16.h>
#include <math.h>

typedef _Float16 v16h __attribute__((ext_vector_type(16)));
typedef float    v8f  __attribute__((ext_vector_type(8)));

#define TILE_N 128   // N columns per block (8 wave32 x 16)

// ---------------------------------------------------------------------------
// concat along channels: out[b, 0:Ch] = A, out[b, Ch:2Ch] = B
// ---------------------------------------------------------------------------
__global__ void concat2_kernel(const float* __restrict__ A,
                               const float* __restrict__ Bp,
                               float* __restrict__ out,
                               int Bn, int Ch, int HW) {
    int idx = blockIdx.x * blockDim.x + threadIdx.x;
    int total = Bn * 2 * Ch * HW;
    if (idx >= total) return;
    int hw = idx % HW;
    int c  = (idx / HW) % (2 * Ch);
    int b  = idx / (HW * 2 * Ch);
    float v = (c < Ch) ? A[((size_t)b * Ch + c) * HW + hw]
                       : Bp[((size_t)b * Ch + (c - Ch)) * HW + hw];
    out[idx] = v;
}

// ---------------------------------------------------------------------------
// Fused QKV 1x1-conv GEMM:
//   {Q,K,V}[b,o,n] = sum_c W{q,k,v}[o,c] * relu(X[b,c,n]) + b{q,k,v}[o]
// Block = 256 threads (8 wave32) -> one 16(M) x 128(N) strip.
// The 32(K) x 128(N) activation panel is staged in LDS once per K-chunk
// (coalesced fp32 loads, fused ReLU + f32->f16), then each wave issues
// three v_wmma_f32_16x16x32_f16 (q, k, v) against the shared B fragment.
// ISA 7.12.2 16-bit fragment packing:
//   lane L: M (A) / N (B) = L & 15 ; element e <-> K = ((e&8)<<1)+(e&7)+8*(L>>4)
// ---------------------------------------------------------------------------
__global__ __launch_bounds__(256)
void conv1x1_qkv_wmma_kernel(const float* __restrict__ X,
                             const float* __restrict__ Wq,
                             const float* __restrict__ Wk,
                             const float* __restrict__ Wv,
                             const float* __restrict__ bq,
                             const float* __restrict__ bk,
                             const float* __restrict__ bv,
                             float* __restrict__ Q,
                             float* __restrict__ Ko,
                             float* __restrict__ V,
                             int Bn, int C, int HW) {
    const int NTG = HW / TILE_N;      // N tile-groups per batch image
    const int MT  = C >> 4;           // M tiles (Co == C for these layers)
    int bid = blockIdx.x;
    int b   = bid / (MT * NTG);
    int rem = bid % (MT * NTG);
    int mt  = rem / NTG;
    int ng  = rem % NTG;
    int n0  = ng * TILE_N;

    int tid  = threadIdx.x;
    int wave = tid >> 5;
    int lane = tid & 31;
    int half = lane >> 4;
    int l16  = lane & 15;

    __shared__ _Float16 sB[32][TILE_N + 8];   // +8 f16 pad against bank conflicts

    const float* Xb = X + (size_t)b * C * HW;
    int m  = mt * 16 + l16;            // weight row this lane packs for A
    int nw = n0 + wave * 16 + l16;     // this lane's output column

    v8f accq = {0.f,0.f,0.f,0.f,0.f,0.f,0.f,0.f};
    v8f acck = accq;
    v8f accv = accq;

    for (int k0 = 0; k0 < C; k0 += 32) {
        __syncthreads();
        // cooperative stage of X[k0:k0+32, n0:n0+128] with fused ReLU
        #pragma unroll
        for (int t = 0; t < 16; ++t) {
            int i  = tid + t * 256;    // 0..4095
            int kk = i >> 7;           // 0..31
            int nn = i & 127;          // 0..127
            float xv = Xb[(size_t)(k0 + kk) * HW + n0 + nn];
            sB[kk][nn] = (_Float16)(xv > 0.f ? xv : 0.f);
        }
        if (k0 + 32 < C) {
            __builtin_prefetch(&Xb[(size_t)(k0 + 32 + (tid >> 5)) * HW + n0 + (tid & 127)], 0, 1);
        }
        __syncthreads();

        // shared B fragment out of LDS
        v16h bfrag;
        #pragma unroll
        for (int e = 0; e < 16; ++e) {
            int kke = ((e & 8) << 1) + (e & 7) + (half << 3);
            bfrag[e] = sB[kke][wave * 16 + l16];
        }
        // three A fragments (weights are tiny and L0-resident)
        v16h aq, ak, av;
        #pragma unroll
        for (int e = 0; e < 16; ++e) {
            int kk = k0 + (((e & 8) << 1) + (e & 7) + (half << 3));
            aq[e] = (_Float16)Wq[(size_t)m * C + kk];
            ak[e] = (_Float16)Wk[(size_t)m * C + kk];
            av[e] = (_Float16)Wv[(size_t)m * C + kk];
        }
        accq = __builtin_amdgcn_wmma_f32_16x16x32_f16(false, aq, false, bfrag,
                                                      (short)0, accq, false, false);
        acck = __builtin_amdgcn_wmma_f32_16x16x32_f16(false, ak, false, bfrag,
                                                      (short)0, acck, false, false);
        accv = __builtin_amdgcn_wmma_f32_16x16x32_f16(false, av, false, bfrag,
                                                      (short)0, accv, false, false);
    }

    size_t obase = (size_t)b * C * HW;
    #pragma unroll
    for (int j = 0; j < 8; ++j) {
        int row  = mt * 16 + j + half * 8;   // C/D layout: VGPR j, lane half
        size_t o = obase + (size_t)row * HW + nw;
        Q[o]  = accq[j] + bq[row];
        Ko[o] = acck[j] + bk[row];
        V[o]  = accv[j] + bv[row];
    }
}

// ---------------------------------------------------------------------------
// Window attention: softmax over 9 taps of q*(k+rel), weighted V sum.
// Out-of-bounds taps of the padded conv equal the conv bias (bk/bv).
// Optional fused residual; strided output (writes into a channel slice).
// ---------------------------------------------------------------------------
__global__ void attn_window_kernel(const float* __restrict__ Q,
                                   const float* __restrict__ Kt,
                                   const float* __restrict__ Vt,
                                   const float* __restrict__ bk,
                                   const float* __restrict__ bv,
                                   const float* __restrict__ relh,
                                   const float* __restrict__ relw,
                                   const float* __restrict__ residual,
                                   float* __restrict__ out,
                                   int Bn, int C, int H, int W,
                                   int outBatchStride, int outChanOff) {
    int idx = blockIdx.x * blockDim.x + threadIdx.x;
    int HW = H * W;
    int total = Bn * C * HW;
    if (idx >= total) return;
    int w = idx % W;
    int h = (idx / W) % H;
    int c = (idx / HW) % C;
    int b = idx / (C * HW);

    float q = Q[idx];
    int halfC = C >> 1;
    const float* Kb = Kt + ((size_t)b * C + c) * HW;
    const float* Vb = Vt + ((size_t)b * C + c) * HW;
    float bkc = bk[c], bvc = bv[c];

    float lg[9], vv[9];
    #pragma unroll
    for (int kh = 0; kh < 3; ++kh) {
        #pragma unroll
        for (int kw = 0; kw < 3; ++kw) {
            int i  = kh * 3 + kw;
            int hh = h + kh - 1;
            int ww = w + kw - 1;
            float kval, vval;
            if (hh >= 0 && hh < H && ww >= 0 && ww < W) {
                kval = Kb[hh * W + ww];
                vval = Vb[hh * W + ww];
            } else {                 // padded-border conv value == bias
                kval = bkc;
                vval = bvc;
            }
            float rb = (c < halfC) ? relh[kh] : relw[kw];
            lg[i] = q * (kval + rb);
            vv[i] = vval;
        }
    }
    float m = lg[0];
    #pragma unroll
    for (int i = 1; i < 9; ++i) m = fmaxf(m, lg[i]);
    float s = 0.f, r = 0.f;
    #pragma unroll
    for (int i = 0; i < 9; ++i) {
        float e = __expf(lg[i] - m);
        s += e;
        r += e * vv[i];
    }
    r /= s;
    if (residual) r += residual[idx];
    out[(size_t)b * outBatchStride + (size_t)(c + outChanOff) * HW + h * W + w] = r;
}

// ---------------------------------------------------------------------------
// 3x3 conv, reflect padding, 1 output channel (depth head)
// ---------------------------------------------------------------------------
__global__ void conv3x3_depth_reflect_kernel(const float* __restrict__ X,
                                             const float* __restrict__ Wc,
                                             const float* __restrict__ bc,
                                             float* __restrict__ out,
                                             int Bn, int C, int H, int W) {
    int idx = blockIdx.x * blockDim.x + threadIdx.x;
    int HW = H * W;
    int total = Bn * HW;
    if (idx >= total) return;
    int w = idx % W;
    int h = (idx / W) % H;
    int b = idx / HW;
    const float* Xb = X + (size_t)b * C * HW;
    float s = bc[0];
    for (int ci = 0; ci < C; ++ci) {
        const float* Xc = Xb + (size_t)ci * HW;
        const float* Wk = Wc + ci * 9;
        #pragma unroll
        for (int dh = 0; dh < 3; ++dh) {
            int hp = h + dh - 1;
            if (hp < 0) hp = -hp;                 // reflect (pad=1)
            if (hp >= H) hp = 2 * H - 2 - hp;
            #pragma unroll
            for (int dw = 0; dw < 3; ++dw) {
                int wp = w + dw - 1;
                if (wp < 0) wp = -wp;
                if (wp >= W) wp = 2 * W - 2 - wp;
                s += Wk[dh * 3 + dw] * Xc[hp * W + wp];
            }
        }
    }
    out[idx] = s;
}

// ---------------------------------------------------------------------------
// 3x3 conv (zero pad, 4 out channels) + tanh + pixel_shuffle(r=2)
// outUp is [B, 1, 2H, 2W]; channel oc -> sub-pixel (oc>>1, oc&1)
// ---------------------------------------------------------------------------
__global__ void conv3x3_up_tanh_shuffle_kernel(const float* __restrict__ X,
                                               const float* __restrict__ Wu,
                                               const float* __restrict__ bu,
                                               float* __restrict__ outUp,
                                               int Bn, int C, int H, int W) {
    int idx = blockIdx.x * blockDim.x + threadIdx.x;
    int HW = H * W;
    int total = Bn * 4 * HW;
    if (idx >= total) return;
    int w  = idx % W;
    int h  = (idx / W) % H;
    int oc = (idx / HW) % 4;
    int b  = idx / (4 * HW);
    const float* Xb = X + (size_t)b * C * HW;
    float s = bu[oc];
    for (int ci = 0; ci < C; ++ci) {
        const float* Xc = Xb + (size_t)ci * HW;
        const float* Wk = Wu + ((size_t)oc * C + ci) * 9;
        #pragma unroll
        for (int dh = 0; dh < 3; ++dh) {
            int hp = h + dh - 1;
            if (hp < 0 || hp >= H) continue;
            #pragma unroll
            for (int dw = 0; dw < 3; ++dw) {
                int wp = w + dw - 1;
                if (wp < 0 || wp >= W) continue;
                s += Wk[dh * 3 + dw] * Xc[hp * W + wp];
            }
        }
    }
    float t = tanhf(s);
    int W2 = 2 * W;
    outUp[(size_t)b * (4 * HW) + (size_t)(2 * h + (oc >> 1)) * W2 + (2 * w + (oc & 1))] = t;
}

// ---------------------------------------------------------------------------
// Host-side orchestration
// ---------------------------------------------------------------------------
namespace {

struct AttnP {
    const float *Wq, *Wk, *Wv, *bq, *bk, *bv, *relh, *relw;
};

inline AttnP attn_from(void* const* d_in, int base) {
    AttnP p;
    p.Wq   = (const float*)d_in[base + 0];
    p.Wk   = (const float*)d_in[base + 1];
    p.Wv   = (const float*)d_in[base + 2];
    p.bq   = (const float*)d_in[base + 3];
    p.bk   = (const float*)d_in[base + 4];
    p.bv   = (const float*)d_in[base + 5];
    p.relh = (const float*)d_in[base + 6];
    p.relw = (const float*)d_in[base + 7];
    return p;
}

inline void launch_attn(const float* x, const AttnP& p,
                        float* Qb, float* Kb, float* Vb,
                        const float* residual, float* out,
                        int Bn, int C, int H, int W,
                        int outBatchStride, int outChanOff,
                        hipStream_t stream) {
    int HW = H * W;
    int blocks = Bn * (C >> 4) * (HW / TILE_N);   // one 16x128 strip per block
    conv1x1_qkv_wmma_kernel<<<blocks, 256, 0, stream>>>(
        x, p.Wq, p.Wk, p.Wv, p.bq, p.bk, p.bv, Qb, Kb, Vb, Bn, C, HW);
    int total = Bn * C * HW;
    attn_window_kernel<<<(total + 255) / 256, 256, 0, stream>>>(
        Qb, Kb, Vb, p.bk, p.bv, p.relh, p.relw, residual, out,
        Bn, C, H, W, outBatchStride, outChanOff);
}

// RAU: out[slice] = attn2(relu(attn1(relu(x)))) + x
inline void launch_rau(const float* x, const AttnP& p1, const AttnP& p2,
                       float* Qb, float* Kb, float* Vb, float* T1,
                       float* out, int outBatchStride, int outChanOff,
                       int Bn, int C, int H, int W, hipStream_t stream) {
    int HW = H * W;
    launch_attn(x, p1, Qb, Kb, Vb, /*residual=*/nullptr, T1,
                Bn, C, H, W, /*stride=*/C * HW, /*off=*/0, stream);
    launch_attn(T1, p2, Qb, Kb, Vb, /*residual=*/x, out,
                Bn, C, H, W, outBatchStride, outChanOff, stream);
}

} // namespace

extern "C" void kernel_launch(void* const* d_in, const int* in_sizes, int n_in,
                              void* d_out, int out_size, void* d_ws, size_t ws_size,
                              hipStream_t stream) {
    (void)in_sizes; (void)n_in; (void)out_size; (void)ws_size;

    const int B = 4, Ch = 32, H = 64, W = 64;
    const int HW = H * W;                 // 4096
    const int F = 2 * Ch;                 // 64
    const int F2 = 2 * F;                 // 128

    const float* dt   = (const float*)d_in[0];
    const float* upt  = (const float*)d_in[1];
    const float* dt_1 = (const float*)d_in[2];
    const float* dt_2 = (const float*)d_in[3];
    AttnP rau1_a1 = attn_from(d_in, 4);
    AttnP rau1_a2 = attn_from(d_in, 12);
    AttnP rau2_a1 = attn_from(d_in, 20);
    AttnP rau2_a2 = attn_from(d_in, 28);
    AttnP rau3_a1 = attn_from(d_in, 36);
    AttnP rau3_a2 = attn_from(d_in, 44);
    const float* conv3_W = (const float*)d_in[52];
    const float* conv3_b = (const float*)d_in[53];
    const float* up_W    = (const float*)d_in[54];
    const float* up_b    = (const float*)d_in[55];

    // workspace carve-out (floats); working set ~56 MB -> L2-resident on MI455X
    float* ws = (float*)d_ws;
    const size_t SZ64  = (size_t)B * F  * HW;   // 1,048,576
    const size_t SZ128 = (size_t)B * F2 * HW;   // 2,097,152
    float* X1 = ws;                 // [B,64,HW]
    float* X2 = X1 + SZ64;          // [B,64,HW]
    float* X3 = X2 + SZ64;          // [B,128,HW]  (RAU1/RAU2 write halves)
    float* Yb = X3 + SZ128;         // [B,128,HW]  RAU3 output
    float* Qb = Yb + SZ128;         // scratch (max 128 ch)
    float* Kb = Qb + SZ128;
    float* Vb = Kb + SZ128;
    float* T1 = Vb + SZ128;

    // 1) channel concats
    {
        int total = B * F * HW;
        concat2_kernel<<<(total + 255) / 256, 256, 0, stream>>>(dt, upt, X1, B, Ch, HW);
        concat2_kernel<<<(total + 255) / 256, 256, 0, stream>>>(dt_1, dt_2, X2, B, Ch, HW);
    }

    // 2) RAU1 -> X3[:, 0:64], RAU2 -> X3[:, 64:128]
    launch_rau(X1, rau1_a1, rau1_a2, Qb, Kb, Vb, T1,
               X3, /*stride=*/F2 * HW, /*off=*/0, B, F, H, W, stream);
    launch_rau(X2, rau2_a1, rau2_a2, Qb, Kb, Vb, T1,
               X3, /*stride=*/F2 * HW, /*off=*/F, B, F, H, W, stream);

    // 3) RAU3 at 128 channels
    launch_rau(X3, rau3_a1, rau3_a2, Qb, Kb, Vb, T1,
               Yb, /*stride=*/F2 * HW, /*off=*/0, B, F2, H, W, stream);

    // 4) heads
    float* out_depth = (float*)d_out;                 // [4,1,64,64]
    float* out_up    = out_depth + (size_t)B * HW;    // [4,1,128,128]
    {
        int total = B * HW;
        conv3x3_depth_reflect_kernel<<<(total + 255) / 256, 256, 0, stream>>>(
            Yb, conv3_W, conv3_b, out_depth, B, F2, H, W);
    }
    {
        int total = B * 4 * HW;
        conv3x3_up_tanh_shuffle_kernel<<<(total + 255) / 256, 256, 0, stream>>>(
            Yb, up_W, up_b, out_up, B, F2, H, W);
    }
}
